// FusedMLPSoftplusWN_69286412419487
// MI455X (gfx1250) — matrix operand order repaired and checked by reading, not compiled, not run on hardware
//
#include <hip/hip_runtime.h>
#include <hip/hip_bf16.h>

// ---------------------------------------------------------------------------
// Fused 3-layer weight-normalized MLP with softplus activations, bf16 WMMA.
//   h1 = softplus(x @ w1); h2 = softplus(h1 @ w2); out = h2 @ w3
// Per-64-row block: x/h1/h2 live in LDS (320 KB/WGP), all matmuls via
// v_wmma_f32_16x16x32_bf16 with 2x2 register macro-tiles per wave (A/B
// fragment reuse), weights pre-normalized + transposed to bf16 in workspace.
// ---------------------------------------------------------------------------

typedef __attribute__((ext_vector_type(16))) __bf16 bf16x16;
typedef __attribute__((ext_vector_type(8)))  float  f32x8;

constexpr int Mdim = 32768, K1 = 512, N1 = 1024, N2 = 1024, N3 = 512;
constexpr int BM = 64;  // rows per workgroup

union FragB16 { bf16x16 v; uint4 q[2]; };

// A-matrix 16x32 bf16 fragment from LDS (row-major [rows][ldk]).
// ISA 7.12.2: lane l holds row (l%16); low lanes K=0..7 & 16..23,
// high lanes K=8..15 & 24..31 -> two 16B chunks at +8*(l>>4), +16+8*(l>>4).
__device__ __forceinline__ bf16x16 load_frag_a_lds(const __bf16* base, int row,
                                                   int ldk, int k0, int lane) {
  FragB16 f;
  const __bf16* p = base + (size_t)row * ldk + k0 + 8 * (lane >> 4);
  f.q[0] = *(const uint4*)p;
  f.q[1] = *(const uint4*)(p + 16);
  return f.v;
}

// B-matrix 32x16 bf16 fragment from global, weights stored transposed [N][K].
// Lane l holds column (l%16), K at k0 + 8*(l>>4) + 0..7 and +16 + 0..7:
// two 16B chunks -> two b128 loads.
__device__ __forceinline__ bf16x16 load_frag_b(const __bf16* wt, int n0,
                                               int ldk, int k0, int lane) {
  FragB16 f;
  const __bf16* p = wt + (size_t)(n0 + (lane & 15)) * ldk + k0 + 8 * (lane >> 4);
  f.q[0] = *(const uint4*)p;
  f.q[1] = *(const uint4*)(p + 16);
  return f.v;
}

#define WMMA_BF16(A, B, C) \
  __builtin_amdgcn_wmma_f32_16x16x32_bf16(false, (A), false, (B), (short)0, \
                                          (C), false, false)

// 32x32 output macro-tile per wave: 2x2 of 16x16 tiles. Each A fragment feeds
// 2 WMMAs and each B fragment feeds 2 WMMAs (halves LDS + L2 traffic/FLOP),
// and the 4 WMMAs per k-step are independent (no RAW hazards).
__device__ __forceinline__ void macro_gemm_2x2(const __bf16* a, int aRow0,
                                               int ldk, int kTot,
                                               const __bf16* wt, int n0,
                                               int lane, f32x8 acc[2][2]) {
  #pragma unroll 2
  for (int k = 0; k < kTot; k += 32) {
    bf16x16 A0 = load_frag_a_lds(a, aRow0 + (lane & 15), ldk, k, lane);
    bf16x16 A1 = load_frag_a_lds(a, aRow0 + 16 + (lane & 15), ldk, k, lane);
    bf16x16 B0 = load_frag_b(wt, n0, kTot, k, lane);
    bf16x16 B1 = load_frag_b(wt, n0 + 16, kTot, k, lane);
    acc[0][0] = WMMA_BF16(A0, B0, acc[0][0]);
    acc[0][1] = WMMA_BF16(A0, B1, acc[0][1]);
    acc[1][0] = WMMA_BF16(A1, B0, acc[1][0]);
    acc[1][1] = WMMA_BF16(A1, B1, acc[1][1]);
  }
}

__device__ __forceinline__ float softplus_f(float a) {
  // Fast stable softplus: max(a,0) + log(1 + exp(-|a|)); result is truncated
  // to bf16 (8 mantissa bits), so HW v_exp/v_log transcendentals suffice.
  return fmaxf(a, 0.0f) + __logf(1.0f + __expf(-fabsf(a)));
}

// C-tile layout (ISA 7.12.2): VGPR r -> row r + 8*(lane>>4), col lane&15.
__device__ __forceinline__ void store_act_lds(__bf16* dst, int ldn, int row0,
                                              int col0, int lane,
                                              const f32x8& acc) {
  const int col   = col0 + (lane & 15);
  const int rbase = row0 + 8 * (lane >> 4);
  #pragma unroll
  for (int r = 0; r < 8; ++r)
    dst[(size_t)(rbase + r) * ldn + col] = (__bf16)softplus_f(acc[r]);
}

// ---------------------------------------------------------------------------
// Prologue: weight-norm each column of v[K][N] (scale g[n]/||v[:,n]||),
// convert to bf16, store TRANSPOSED as wt[N][K]. One block per column.
// ---------------------------------------------------------------------------
__global__ void wn_transpose_kernel(const float* __restrict__ v,
                                    const float* __restrict__ g,
                                    __bf16* __restrict__ wt, int K, int N) {
  __shared__ float red[256];
  const int n = blockIdx.x;
  float s = 0.0f;
  for (int k = threadIdx.x; k < K; k += 256) {
    float t = v[(size_t)k * N + n];
    s += t * t;
  }
  red[threadIdx.x] = s;
  __syncthreads();
  for (int w = 128; w > 0; w >>= 1) {
    if ((int)threadIdx.x < w) red[threadIdx.x] += red[threadIdx.x + w];
    __syncthreads();
  }
  const float scale = g[n] * rsqrtf(red[0]);
  for (int k = threadIdx.x; k < K; k += 256)
    wt[(size_t)n * K + k] = (__bf16)(v[(size_t)k * N + n] * scale);
}

// ---------------------------------------------------------------------------
// Main fused kernel. 256 threads = 8 wave32s; 64 rows per block.
// LDS: xs 64x512 bf16 (64KB) | h1 64x1024 bf16 (128KB) | h2 64x1024 bf16
// (128KB) = 320KB.
// ---------------------------------------------------------------------------
__launch_bounds__(256, 1)
__global__ void fused_mlp_kernel(const float* __restrict__ x,
                                 const __bf16* __restrict__ w1t,
                                 const __bf16* __restrict__ w2t,
                                 const __bf16* __restrict__ w3t,
                                 float* __restrict__ out) {
  extern __shared__ __align__(16) char smem[];
  __bf16* xs = (__bf16*)smem;                                   // [BM][K1]
  __bf16* h1 = (__bf16*)(smem + (size_t)BM * K1 * 2);           // [BM][N1]
  __bf16* h2 = (__bf16*)(smem + (size_t)BM * (K1 + N1) * 2);    // [BM][N2]

  const int tid  = threadIdx.x;
  const int lane = tid & 31;
  const int wave = tid >> 5;
  const size_t rowBlock = (size_t)blockIdx.x * BM;

  // ---- Phase 0: stage x block, f32 -> bf16, into LDS (vectorized) ----
  {
    const float4* xg = (const float4*)(x + rowBlock * K1);
    unsigned long long* xsq = (unsigned long long*)xs;
    #pragma unroll 2
    for (int i = tid; i < BM * K1 / 4; i += 256) {
      float4 f = xg[i];
      union { __bf16 b[4]; unsigned long long u; } pk;
      pk.b[0] = (__bf16)f.x; pk.b[1] = (__bf16)f.y;
      pk.b[2] = (__bf16)f.z; pk.b[3] = (__bf16)f.w;
      xsq[i] = pk.u;
    }
  }
  __syncthreads();

  // ---- Phase 1: h1 = softplus(x @ w1) : 2 x 32 macro-tiles of 32x32 ----
  for (int t = wave; t < (BM / 32) * (N1 / 32); t += 8) {
    const int rm = t / (N1 / 32), cm = t % (N1 / 32);
    f32x8 acc[2][2] = {};
    macro_gemm_2x2(xs, rm * 32, K1, K1, w1t, cm * 32, lane, acc);
    #pragma unroll
    for (int i = 0; i < 2; ++i)
      #pragma unroll
      for (int j = 0; j < 2; ++j)
        store_act_lds(h1, N1, rm * 32 + i * 16, cm * 32 + j * 16, lane,
                      acc[i][j]);
  }
  __syncthreads();

  // ---- Phase 2: h2 = softplus(h1 @ w2) : 2 x 32 macro-tiles ----
  for (int t = wave; t < (BM / 32) * (N2 / 32); t += 8) {
    const int rm = t / (N2 / 32), cm = t % (N2 / 32);
    f32x8 acc[2][2] = {};
    macro_gemm_2x2(h1, rm * 32, N1, N1, w2t, cm * 32, lane, acc);
    #pragma unroll
    for (int i = 0; i < 2; ++i)
      #pragma unroll
      for (int j = 0; j < 2; ++j)
        store_act_lds(h2, N2, rm * 32 + i * 16, cm * 32 + j * 16, lane,
                      acc[i][j]);
  }
  __syncthreads();

  // ---- Phase 3: out = h2 @ w3 : 2 x 16 macro-tiles, f32 to HBM ----
  for (int t = wave; t < (BM / 32) * (N3 / 32); t += 8) {
    const int rm = t / (N3 / 32), cm = t % (N3 / 32);
    f32x8 acc[2][2] = {};
    macro_gemm_2x2(h2, rm * 32, N2, N2, w3t, cm * 32, lane, acc);
    #pragma unroll
    for (int i = 0; i < 2; ++i)
      #pragma unroll
      for (int j = 0; j < 2; ++j) {
        const int col   = cm * 32 + j * 16 + (lane & 15);
        const int rbase = rm * 32 + i * 16 + 8 * (lane >> 4);
        #pragma unroll
        for (int r = 0; r < 8; ++r)
          out[(rowBlock + rbase + r) * N3 + col] = acc[i][j][r];
      }
  }
}

// ---------------------------------------------------------------------------
extern "C" void kernel_launch(void* const* d_in, const int* in_sizes, int n_in,
                              void* d_out, int out_size, void* d_ws,
                              size_t ws_size, hipStream_t stream) {
  (void)in_sizes; (void)n_in; (void)out_size; (void)ws_size;
  const float* x  = (const float*)d_in[0];
  const float* v1 = (const float*)d_in[1];
  const float* g1 = (const float*)d_in[2];
  const float* v2 = (const float*)d_in[3];
  const float* g2 = (const float*)d_in[4];
  const float* v3 = (const float*)d_in[5];
  const float* g3 = (const float*)d_in[6];
  float* out = (float*)d_out;

  // Workspace: bf16 transposed normalized weights (~4 MB total)
  __bf16* w1t = (__bf16*)d_ws;                       // [N1][K1]
  __bf16* w2t = w1t + (size_t)N1 * K1;               // [N2][N1]
  __bf16* w3t = w2t + (size_t)N2 * N1;               // [N3][N2]

  wn_transpose_kernel<<<N1, 256, 0, stream>>>(v1, g1, w1t, K1, N1);
  wn_transpose_kernel<<<N2, 256, 0, stream>>>(v2, g2, w2t, N1, N2);
  wn_transpose_kernel<<<N3, 256, 0, stream>>>(v3, g3, w3t, N2, N3);

  constexpr size_t smemBytes = (size_t)BM * (K1 + N1 + N2) * sizeof(__bf16); // 320 KiB
  hipFuncSetAttribute(reinterpret_cast<const void*>(fused_mlp_kernel),
                      hipFuncAttributeMaxDynamicSharedMemorySize,
                      (int)smemBytes);
  fused_mlp_kernel<<<Mdim / BM, 256, smemBytes, stream>>>(x, w1t, w2t, w3t, out);
}
